// TransformerBlock_44899588112943
// MI455X (gfx1250) — matrix operand order repaired and checked by reading, not compiled, not run on hardware
//
#include <hip/hip_runtime.h>
#include <cstdint>

#define SEQ   2048
#define BATCH 2
#define NTOK  (SEQ * BATCH)   // 4096 tokens
#define EMB   1024
#define HEADS 16
#define DH    64
#define DFF   4096

typedef __attribute__((ext_vector_type(16))) __bf16 v16bf;
typedef __attribute__((ext_vector_type(8)))  float  v8f;

union FragU { uint4 u[2]; v16bf v; };

__device__ __forceinline__ uint16_t f2bf(float f) {
    uint32_t u = __float_as_uint(f);
    uint32_t r = (u + 0x7FFFu + ((u >> 16) & 1u)) >> 16;   // round-nearest-even
    return (uint16_t)r;
}

// Load a 16-element-per-lane (32 byte) WMMA fragment as two 16B chunks
// separated by `gap` bf16 elements. gap=16 -> A-fragment split; gap=8 -> contiguous.
__device__ __forceinline__ v16bf load_frag(const uint16_t* p, int gap) {
    FragU r;
    r.u[0] = *reinterpret_cast<const uint4*>(p);
    r.u[1] = *reinterpret_cast<const uint4*>(p + gap);
    return r.v;
}

#define WMMA_BF16(A, B, C) \
    __builtin_amdgcn_wmma_f32_16x16x32_bf16(false, (A), false, (B), (short)0, (C), false, false)

// ---- DPP16 16-lane butterfly reductions (pure VALU, no LDS crossbar) -------
template <int CTRL>
__device__ __forceinline__ float dppf(float x) {
    return __int_as_float(__builtin_amdgcn_update_dpp(
        0, __float_as_int(x), CTRL, 0xf, 0xf, true));
}
__device__ __forceinline__ float red16_max(float t) {
    t = fmaxf(t, dppf<0xB1>(t));    // quad_perm(1,0,3,2)  : lane ^ 1
    t = fmaxf(t, dppf<0x4E>(t));    // quad_perm(2,3,0,1)  : lane ^ 2
    t = fmaxf(t, dppf<0x141>(t));   // row_half_mirror     : combines quads in 8
    t = fmaxf(t, dppf<0x140>(t));   // row_mirror          : combines halves in 16
    return t;
}
__device__ __forceinline__ float red16_sum(float t) {
    t += dppf<0xB1>(t);
    t += dppf<0x4E>(t);
    t += dppf<0x141>(t);
    t += dppf<0x140>(t);
    return t;
}

// ---------------------------------------------------------------------------
// Pack f32 weight [K x N] row-major into bf16 WMMA B-fragment blocks.
// Block (kt, nt) covers K rows kt*32..+31, N cols nt*16..+15.
// Lane l owns col n = nt*16+(l&15), K = kt*32+(l>>4)*16+j, 16 consecutive bf16
// (32 bytes) at Wp[(blk*32 + l)*16 + j], blk = nt*(K/32) + kt.
// ---------------------------------------------------------------------------
__global__ __launch_bounds__(256)
void pack_b_kernel(const float* __restrict__ W, uint16_t* __restrict__ Wp,
                   int K, int N) {
    int idx  = blockIdx.x * 256 + threadIdx.x;
    int j    = idx & 15;
    int lane = (idx >> 4) & 31;
    int blk  = idx >> 9;
    int ktiles = K >> 5;
    int kt = blk % ktiles;
    int nt = blk / ktiles;
    int k  = kt * 32 + ((lane >> 4) << 4) + j;
    int n  = nt * 16 + (lane & 15);
    Wp[idx] = f2bf(W[(size_t)k * N + n]);
}

// ---------------------------------------------------------------------------
// Fused (a [+ b]) + LayerNorm -> bf16 rows; optional f32 residual carrier.
// ---------------------------------------------------------------------------
__global__ __launch_bounds__(256)
void add_ln_kernel(const float* __restrict__ a, const float* __restrict__ b,
                   const float* __restrict__ g, const float* __restrict__ be,
                   float* __restrict__ resOut, uint16_t* __restrict__ hbf) {
    __shared__ float s1[256], s2[256];
    const int row = blockIdx.x;
    const int t   = threadIdx.x;
    const float* pa = a + (size_t)row * EMB;
    const float* pb = b ? b + (size_t)row * EMB : nullptr;

    float vals[4], sum = 0.f, sq = 0.f;
#pragma unroll
    for (int e = 0; e < 4; ++e) {
        int i = t + e * 256;
        float x = pa[i] + (pb ? pb[i] : 0.f);
        vals[e] = x;
        if (resOut) resOut[(size_t)row * EMB + i] = x;
        sum += x; sq += x * x;
    }
    s1[t] = sum; s2[t] = sq;
    __syncthreads();
    for (int s = 128; s > 0; s >>= 1) {
        if (t < s) { s1[t] += s1[t + s]; s2[t] += s2[t + s]; }
        __syncthreads();
    }
    float mu   = s1[0] * (1.f / EMB);
    float var  = s2[0] * (1.f / EMB) - mu * mu;
    float rstd = rsqrtf(var + 1e-6f);
#pragma unroll
    for (int e = 0; e < 4; ++e) {
        int i = t + e * 256;
        float o = (vals[e] - mu) * rstd * g[i] + be[i];
        hbf[(size_t)row * EMB + i] = f2bf(o);
    }
}

// ---------------------------------------------------------------------------
// bf16 transpose: v [BATCH][SEQ][EMB] -> vT [BATCH][EMB][SEQ]
// ---------------------------------------------------------------------------
__global__ __launch_bounds__(256)
void transpose_v_kernel(const uint16_t* __restrict__ v, uint16_t* __restrict__ vT) {
    size_t idx = (size_t)blockIdx.x * 256 + threadIdx.x;  // (bb, d, m), m fastest
    int m  = (int)(idx & (SEQ - 1));
    int d  = (int)((idx >> 11) & (EMB - 1));
    int bb = (int)(idx >> 21);
    vT[idx] = v[((size_t)bb * SEQ + m) * EMB + d];
}

// ---------------------------------------------------------------------------
// bf16 WMMA GEMM: C[M,N] = A[M,K] @ B[K,N] + bias. 64x64 wave tile:
// 4x4 accumulators, 16 WMMAs per 8 fragment loads per K=32 step.
// MODE 0: f32 out (+ optional residual); MODE 1: exact GELU -> bf16;
// MODE 2: bf16 out.
// ---------------------------------------------------------------------------
template <int MODE>
__global__ __launch_bounds__(256)
void gemm_wmma_kernel(const uint16_t* __restrict__ A, const uint16_t* __restrict__ Bp,
                      const float* __restrict__ bias, const float* __restrict__ res,
                      void* __restrict__ Cout, int M, int N, int K) {
    const int lane = threadIdx.x & 31;
    const int wid  = (blockIdx.x << 3) + (threadIdx.x >> 5);
    const int tiles_n = N >> 6;
    const int mt = (wid / tiles_n) << 6;
    const int nt = (wid % tiles_n) << 6;
    const int ktiles = K >> 5;
    const int half = lane >> 4;
    const int lrow = lane & 15;

    v8f acc[4][4] = {};

    const uint16_t* aBase = A + (size_t)(mt + lrow) * K + half * 16;
    const uint16_t* bBase = Bp + (size_t)lane * 16;
    size_t bBlk[4];
#pragma unroll
    for (int j = 0; j < 4; ++j) bBlk[j] = (size_t)((nt >> 4) + j) * ktiles * 512;

    for (int kt = 0; kt < ktiles; ++kt) {
        int kpf = (kt + 4 < ktiles) ? kt + 4 : kt;   // prefetch distance: 4 K-tiles
        v16bf a[4], b[4];
#pragma unroll
        for (int i = 0; i < 4; ++i) {
            a[i] = load_frag(aBase + (size_t)(16 * i) * K + kt * 32, 8);
            __builtin_prefetch(aBase + (size_t)(16 * i) * K + kpf * 32, 0, 1);
        }
#pragma unroll
        for (int j = 0; j < 4; ++j) {
            b[j] = load_frag(bBase + bBlk[j] + (size_t)kt * 512, 8);
            __builtin_prefetch(bBase + bBlk[j] + (size_t)kpf * 512, 0, 1);
        }
#pragma unroll
        for (int i = 0; i < 4; ++i)
#pragma unroll
            for (int j = 0; j < 4; ++j)
                acc[i][j] = WMMA_BF16(a[i], b[j], acc[i][j]);
    }

#pragma unroll
    for (int i = 0; i < 4; ++i)
#pragma unroll
        for (int j = 0; j < 4; ++j) {
            int c = nt + j * 16 + lrow;
            float bj = bias[c];
#pragma unroll
            for (int r = 0; r < 8; ++r) {
                int R = mt + i * 16 + r + half * 8;
                float v = acc[i][j][r] + bj;
                size_t off = (size_t)R * N + c;
                if (MODE == 0) {
                    float o = v + (res ? res[off] : 0.f);
                    ((float*)Cout)[off] = o;
                } else if (MODE == 1) {
                    float ge = 0.5f * v * (1.f + erff(v * 0.70710678118654752f));
                    ((uint16_t*)Cout)[off] = f2bf(ge);
                } else {
                    ((uint16_t*)Cout)[off] = f2bf(v);
                }
            }
        }
}

// ---------------------------------------------------------------------------
// Flash-style attention. One wave per (batch, head, 16-query-row tile).
// q/k bf16 [BATCH*SEQ][EMB]; vT bf16 [BATCH][EMB][SEQ]; out f32.
// Reference applies NO 1/sqrt(dh) scale -> neither do we.
// ---------------------------------------------------------------------------
__global__ __launch_bounds__(256)
void attn_wmma_kernel(const uint16_t* __restrict__ q, const uint16_t* __restrict__ k,
                      const uint16_t* __restrict__ vT, float* __restrict__ out) {
    __shared__ uint16_t pbuf[8][16][40];   // per-wave 16x32 p tile, row stride 80B (16B-aligned)
    const int lane = threadIdx.x & 31;
    const int w    = threadIdx.x >> 5;
    const int wid  = (blockIdx.x << 3) + w;       // [0, 4096)
    const int qt   = wid & 127;
    const int h    = (wid >> 7) & 15;
    const int bb   = wid >> 11;
    const int half = lane >> 4;
    const int lrow = lane & 15;

    // persistent q fragments (dh = 64 -> 2 K-chunks of 32)
    const size_t qOff = ((size_t)(bb * SEQ + qt * 16 + lrow)) * EMB + h * DH + half * 16;
    v16bf qa0 = load_frag(q + qOff,      8);
    v16bf qa1 = load_frag(q + qOff + 32, 8);

    v8f acc[4] = {};
    float mrow[8], lsum[8];
#pragma unroll
    for (int r = 0; r < 8; ++r) { mrow[r] = -1e30f; lsum[r] = 0.f; }

    for (int m0 = 0; m0 < SEQ; m0 += 32) {
        // scores: B[d, m] = k[m, d] -> contiguous d per lane from k rows
        const uint16_t* kb_lo = k + ((size_t)(bb * SEQ + m0 + lrow)) * EMB + h * DH + half * 16;
        const uint16_t* kb_hi = kb_lo + (size_t)16 * EMB;
        // prefetch next key tile's rows (speculative; dropped if OOB)
        if (m0 + 32 < SEQ) {
            __builtin_prefetch(kb_lo + (size_t)32 * EMB, 0, 1);
            __builtin_prefetch(kb_hi + (size_t)32 * EMB, 0, 1);
        }
        v16bf b00 = load_frag(kb_lo,      8);   // d 0..31,  keys m0..m0+15
        v16bf b01 = load_frag(kb_lo + 32, 8);   // d 32..63
        v16bf b10 = load_frag(kb_hi,      8);   // keys m0+16..m0+31
        v16bf b11 = load_frag(kb_hi + 32, 8);
        v8f s0 = {}, s1 = {};
        s0 = WMMA_BF16(qa0, b00, s0);
        s0 = WMMA_BF16(qa1, b01, s0);
        s1 = WMMA_BF16(qa0, b10, s1);
        s1 = WMMA_BF16(qa1, b11, s1);

        // online softmax over these 32 key columns (DPP16 butterfly reductions)
#pragma unroll
        for (int r = 0; r < 8; ++r) {
            float t = red16_max(fmaxf(s0[r], s1[r]));
            float mn = fmaxf(mrow[r], t);
            float sc = __expf(mrow[r] - mn);
            float e0 = __expf(s0[r] - mn);
            float e1 = __expf(s1[r] - mn);
            float rs = red16_sum(e0 + e1);
            lsum[r] = lsum[r] * sc + rs;
            mrow[r] = mn;
            acc[0][r] *= sc; acc[1][r] *= sc; acc[2][r] *= sc; acc[3][r] *= sc;
            pbuf[w][r + half * 8][lrow]      = f2bf(e0);
            pbuf[w][r + half * 8][16 + lrow] = f2bf(e1);
        }
        // C-layout -> A-layout re-tiling through per-wave LDS (in-order DS within wave)
        asm volatile("s_wait_dscnt 0" ::: "memory");
        v16bf pfrag = load_frag(&pbuf[w][lrow][half * 16], 8);

        // out += p @ v : B[m, d] = vT[d, m] -> contiguous m per lane from vT rows
#pragma unroll
        for (int dt = 0; dt < 4; ++dt) {
            const uint16_t* vb =
                vT + ((size_t)(bb * EMB + h * DH + dt * 16 + lrow)) * SEQ + m0 + half * 16;
            v16bf bv = load_frag(vb, 8);
            acc[dt] = WMMA_BF16(pfrag, bv, acc[dt]);
        }
    }

#pragma unroll
    for (int r = 0; r < 8; ++r) {
        float inv = 1.f / lsum[r];
        int R = bb * SEQ + qt * 16 + r + half * 8;
#pragma unroll
        for (int dt = 0; dt < 4; ++dt)
            out[(size_t)R * EMB + h * DH + dt * 16 + lrow] = acc[dt][r] * inv;
    }
}

// ---------------------------------------------------------------------------
extern "C" void kernel_launch(void* const* d_in, const int* in_sizes, int n_in,
                              void* d_out, int out_size, void* d_ws, size_t ws_size,
                              hipStream_t stream) {
    (void)in_sizes; (void)n_in; (void)out_size; (void)ws_size;
    const float* x   = (const float*)d_in[0];
    const float* Wq  = (const float*)d_in[1];
    const float* bq  = (const float*)d_in[2];
    const float* Wk  = (const float*)d_in[3];
    const float* bk  = (const float*)d_in[4];
    const float* Wv  = (const float*)d_in[5];
    const float* bv  = (const float*)d_in[6];
    const float* g1  = (const float*)d_in[7];
    const float* be1 = (const float*)d_in[8];
    const float* g2  = (const float*)d_in[9];
    const float* be2 = (const float*)d_in[10];
    const float* W1  = (const float*)d_in[11];
    const float* b1  = (const float*)d_in[12];
    const float* W2  = (const float*)d_in[13];
    const float* b2  = (const float*)d_in[14];

    char* ws = (char*)d_ws;
    size_t off = 0;
    auto alloc = [&](size_t bytes) -> char* {
        char* p = ws + off;
        off = (off + bytes + 255) & ~(size_t)255;
        return p;
    };
    uint16_t* hbf  = (uint16_t*)alloc((size_t)NTOK * EMB * 2);
    uint16_t* qbf  = (uint16_t*)alloc((size_t)NTOK * EMB * 2);
    uint16_t* kbf  = (uint16_t*)alloc((size_t)NTOK * EMB * 2);
    uint16_t* vbf  = (uint16_t*)alloc((size_t)NTOK * EMB * 2);
    uint16_t* vTbf = (uint16_t*)alloc((size_t)NTOK * EMB * 2);
    uint16_t* h2bf = (uint16_t*)alloc((size_t)NTOK * EMB * 2);
    uint16_t* midbf= (uint16_t*)alloc((size_t)NTOK * DFF * 2);
    uint16_t* Wqp  = (uint16_t*)alloc((size_t)EMB * EMB * 2);
    uint16_t* Wkp  = (uint16_t*)alloc((size_t)EMB * EMB * 2);
    uint16_t* Wvp  = (uint16_t*)alloc((size_t)EMB * EMB * 2);
    uint16_t* W1p  = (uint16_t*)alloc((size_t)EMB * DFF * 2);
    uint16_t* W2p  = (uint16_t*)alloc((size_t)DFF * EMB * 2);
    float*    attn = (float*)   alloc((size_t)NTOK * EMB * 4);
    float*    res2 = (float*)   alloc((size_t)NTOK * EMB * 4);

    // 1) pack weights to bf16 WMMA-fragment layout
    pack_b_kernel<<<(EMB * EMB) / 256, 256, 0, stream>>>(Wq, Wqp, EMB, EMB);
    pack_b_kernel<<<(EMB * EMB) / 256, 256, 0, stream>>>(Wk, Wkp, EMB, EMB);
    pack_b_kernel<<<(EMB * EMB) / 256, 256, 0, stream>>>(Wv, Wvp, EMB, EMB);
    pack_b_kernel<<<(EMB * DFF) / 256, 256, 0, stream>>>(W1, W1p, EMB, DFF);
    pack_b_kernel<<<(DFF * EMB) / 256, 256, 0, stream>>>(W2, W2p, DFF, EMB);

    // 2) LN1 -> bf16 activations
    add_ln_kernel<<<NTOK, 256, 0, stream>>>(x, nullptr, g1, be1, nullptr, hbf);

    // 3) QKV projections (bf16 out); 64x64 wave tiles
    const int qkvBlocks = ((NTOK / 64) * (EMB / 64)) / 8;
    gemm_wmma_kernel<2><<<qkvBlocks, 256, 0, stream>>>(hbf, Wqp, bq, nullptr, qbf, NTOK, EMB, EMB);
    gemm_wmma_kernel<2><<<qkvBlocks, 256, 0, stream>>>(hbf, Wkp, bk, nullptr, kbf, NTOK, EMB, EMB);
    gemm_wmma_kernel<2><<<qkvBlocks, 256, 0, stream>>>(hbf, Wvp, bv, nullptr, vbf, NTOK, EMB, EMB);

    // 4) transpose v for contiguous PV fragments
    transpose_v_kernel<<<((size_t)NTOK * EMB) / 256, 256, 0, stream>>>(vbf, vTbf);

    // 5) flash attention (4096 wave-tasks, 8 waves/block)
    attn_wmma_kernel<<<512, 256, 0, stream>>>(qbf, kbf, vTbf, attn);

    // 6) residual + LN2 (keep f32 residual, bf16 normalized)
    add_ln_kernel<<<NTOK, 256, 0, stream>>>(attn, x, g2, be2, res2, h2bf);

    // 7) FFN up + exact GELU (bf16 out)
    const int ffn1Blocks = ((NTOK / 64) * (DFF / 64)) / 8;
    gemm_wmma_kernel<1><<<ffn1Blocks, 256, 0, stream>>>(h2bf, W1p, b1, nullptr, midbf, NTOK, DFF, EMB);

    // 8) FFN down + residual -> f32 final output
    const int ffn2Blocks = ((NTOK / 64) * (EMB / 64)) / 8;
    gemm_wmma_kernel<0><<<ffn2Blocks, 256, 0, stream>>>(midbf, W2p, b2, res2, (float*)d_out, NTOK, EMB, DFF);
}
